// CrossAttentionBlock_4552665334337
// MI455X (gfx1250) — matrix operand order repaired and checked by reading, not compiled, not run on hardware
//
#include <hip/hip_runtime.h>
#include <stdint.h>

#define DIM 1024
#define NH 16
#define HD 64
#define NB 8
#define LQ 512
#define LK 1024
#define EPS 1e-5f

typedef __attribute__((ext_vector_type(16))) _Float16 v16h;
typedef __attribute__((ext_vector_type(8)))  _Float16 v8h;
typedef __attribute__((ext_vector_type(8)))  float    v8f;

// ---------------------------------------------------------------------------
// WMMA fragment loaders for V_WMMA_F32_16X16X32_F16 (ISA 7.12.2 layouts).
// A 16x32 f16: lane l holds row (l&15); K elems e<8 -> K=e+(l>>4)*8,
//              e>=8 -> K=e+8+(l>>4)*8  (two contiguous 8-half chunks).
// B 32x16 f16: lane l holds col (l&15); K elem e -> same K as A elem e
//              (two contiguous 8-half chunks when source is K-contiguous).
// C/D 16x16 f32: lane l col (l&15), VGPR r -> row r+(l>>4)*8.
// ---------------------------------------------------------------------------

static __device__ __forceinline__ v16h
load_a_rowmajor_f16(const _Float16* base, int lda, int row0, int k0, int lane) {
    const _Float16* p = base + (size_t)(row0 + (lane & 15)) * lda + k0 + (lane >> 4) * 8;
    v8h lo = *(const v8h*)p;
    v8h hi = *(const v8h*)(p + 16);
    v16h a;
#pragma unroll
    for (int i = 0; i < 8; ++i) { a[i] = lo[i]; a[i + 8] = hi[i]; }
    return a;
}

// B fragment where memory is [n][k] row-major (X @ W^T, K^T, V^T): K contiguous.
static __device__ __forceinline__ v16h
load_b_ncontig_k(const _Float16* base, int ldb, int col0, int k0, int lane) {
    const _Float16* p = base + (size_t)(col0 + (lane & 15)) * ldb + k0 + (lane >> 4) * 8;
    v8h lo = *(const v8h*)p;
    v8h hi = *(const v8h*)(p + 16);
    v16h b;
#pragma unroll
    for (int i = 0; i < 8; ++i) { b[i] = lo[i]; b[i + 8] = hi[i]; }
    return b;
}

// A fragment from f32 LDS tile (probs), converting to f16.
static __device__ __forceinline__ v16h
load_a_lds_f32(const float* s, int stride, int row0, int k0, int lane) {
    const float* p = s + (size_t)(row0 + (lane & 15)) * stride + k0 + (lane >> 4) * 8;
    v16h a;
#pragma unroll
    for (int i = 0; i < 8; ++i) { a[i] = (_Float16)p[i]; a[i + 8] = (_Float16)p[i + 16]; }
    return a;
}

static __device__ __forceinline__ v8f
wmma16x16x32(v16h a, v16h b, v8f c) {
    return __builtin_amdgcn_wmma_f32_16x16x32_f16(false, a, false, b, (short)0, c, false, false);
}

// ---------------------------------------------------------------------------
// Async global->LDS staging (gfx1250 GLOBAL_LOAD_ASYNC_TO_LDS_B128, ASYNCcnt).
// LDS tile: 128 rows x 32 halfs, padded to 40 halfs (80 B) per row so both
// A-fragment (16 rows) and B-fragment (16 cols) reads hit distinct banks.
// ---------------------------------------------------------------------------
#define TP 40          // padded LDS row pitch in halfs
#define KSTEP 32

static __device__ __forceinline__ void
async_chunk(const _Float16* __restrict__ g, int ld, int row0, int k0,
            _Float16* lbase, int c) {
    int row = c >> 2;
    int ch  = (c & 3) * 8;                         // half offset within row
    uint32_t lds = (uint32_t)(uintptr_t)(lbase + row * TP + ch);
    const void* gp = (const void*)(g + (size_t)(row0 + row) * ld + k0 + ch);
    asm volatile("global_load_async_to_lds_b128 %0, %1, off"
                 :: "v"(lds), "v"(gp) : "memory");
}

// 128 rows x 64 B = 512 x 16 B chunks; exactly 2 per thread (tid < 256),
// issued straight-line so no EXEC-mask predication appears around the issues.
static __device__ __forceinline__ void
async_tile_load(const _Float16* __restrict__ g, int ld, int row0, int k0,
                _Float16* lbase, int tid) {
    async_chunk(g, ld, row0, k0, lbase, tid);
    async_chunk(g, ld, row0, k0, lbase, tid + 256);
}

static __device__ __forceinline__ void wait_async0() {
    asm volatile("s_wait_asynccnt 0x0" ::: "memory");
}

// ---------------------------------------------------------------------------
// Kernels
// ---------------------------------------------------------------------------

__global__ void cvt_f32_to_f16(const float* __restrict__ x, _Float16* __restrict__ y, int n) {
    int i = blockIdx.x * blockDim.x + threadIdx.x;
    if (i < n) y[i] = (_Float16)x[i];
}

// v (NB,LK,DIM) -> vT (NB,DIM,LK): vT[b][d][kk] = v[b][kk][d]. 32x32 LDS tiles.
__global__ __launch_bounds__(256)
void transpose_v_kernel(const _Float16* __restrict__ v, _Float16* __restrict__ vT) {
    __shared__ _Float16 tile[32][33];
    int d0  = blockIdx.x * 32;
    int kk0 = blockIdx.y * 32;
    int b   = blockIdx.z;
    int tx = threadIdx.x & 31, ty = threadIdx.x >> 5;
    for (int r = ty; r < 32; r += 8)
        tile[r][tx] = v[((size_t)b * LK + kk0 + r) * DIM + d0 + tx];
    __syncthreads();
    for (int r = ty; r < 32; r += 8)
        vT[((size_t)b * DIM + d0 + r) * LK + kk0 + tx] = tile[tx][r];
}

// Y[m][n] = sum_k X[m][k] * W[n][k] + bias[n], N = K = 1024, f16 out.
// WG 256 thr = 8 waves (2x4); wave 64x32 (4x2 tiles); WG tile 128x128.
// A/B K-step tiles double-buffered in LDS via async copies.
__global__ __launch_bounds__(256)
void gemm_xwT_bias_f16(const _Float16* __restrict__ X, const _Float16* __restrict__ W,
                       const float* __restrict__ bias, _Float16* __restrict__ Y) {
    __shared__ _Float16 As[2][128 * TP];
    __shared__ _Float16 Bs[2][128 * TP];
    int tid = threadIdx.x;
    int lane = tid & 31, wave = tid >> 5;
    int bm = blockIdx.y * 128, bn = blockIdx.x * 128;
    int wm = (wave >> 2) * 64, wn = (wave & 3) * 32;
    v8f acc[4][2] = {};

    async_tile_load(X, DIM, bm, 0, As[0], tid);
    async_tile_load(W, DIM, bn, 0, Bs[0], tid);

    for (int i = 0; i < DIM / KSTEP; ++i) {
        int cur = i & 1;
        wait_async0();
        __syncthreads();                               // tiles visible to all waves
        if (i + 1 < DIM / KSTEP) {                     // prefetch overlaps compute
            async_tile_load(X, DIM, bm, (i + 1) * KSTEP, As[1 - cur], tid);
            async_tile_load(W, DIM, bn, (i + 1) * KSTEP, Bs[1 - cur], tid);
        }
        v16h a[4], b[2];
#pragma unroll
        for (int ii = 0; ii < 4; ++ii) a[ii] = load_a_rowmajor_f16(As[cur], TP, wm + 16 * ii, 0, lane);
#pragma unroll
        for (int j = 0; j < 2; ++j) b[j] = load_b_ncontig_k(Bs[cur], TP, wn + 16 * j, 0, lane);
#pragma unroll
        for (int ii = 0; ii < 4; ++ii)
#pragma unroll
            for (int j = 0; j < 2; ++j)
                acc[ii][j] = wmma16x16x32(a[ii], b[j], acc[ii][j]);
        __syncthreads();                               // readers done before buf reuse
    }

    int m0 = bm + wm, n0 = bn + wn;
#pragma unroll
    for (int ii = 0; ii < 4; ++ii)
#pragma unroll
        for (int j = 0; j < 2; ++j) {
            int col = n0 + 16 * j + (lane & 15);
            float bv = bias[col];
            int rb = m0 + 16 * ii + (lane >> 4) * 8;
#pragma unroll
            for (int r = 0; r < 8; ++r)
                Y[(size_t)(rb + r) * DIM + col] = (_Float16)(acc[ii][j][r] + bv);
        }
}

// attended = ctx @ out_w^T + out_b + residual(text), f32 out. Same staging.
__global__ __launch_bounds__(256)
void gemm_out_residual(const _Float16* __restrict__ X, const _Float16* __restrict__ W,
                       const float* __restrict__ bias, const float* __restrict__ resid,
                       float* __restrict__ Y) {
    __shared__ _Float16 As[2][128 * TP];
    __shared__ _Float16 Bs[2][128 * TP];
    int tid = threadIdx.x;
    int lane = tid & 31, wave = tid >> 5;
    int bm = blockIdx.y * 128, bn = blockIdx.x * 128;
    int wm = (wave >> 2) * 64, wn = (wave & 3) * 32;
    v8f acc[4][2] = {};

    async_tile_load(X, DIM, bm, 0, As[0], tid);
    async_tile_load(W, DIM, bn, 0, Bs[0], tid);

    for (int i = 0; i < DIM / KSTEP; ++i) {
        int cur = i & 1;
        wait_async0();
        __syncthreads();
        if (i + 1 < DIM / KSTEP) {
            async_tile_load(X, DIM, bm, (i + 1) * KSTEP, As[1 - cur], tid);
            async_tile_load(W, DIM, bn, (i + 1) * KSTEP, Bs[1 - cur], tid);
        }
        v16h a[4], b[2];
#pragma unroll
        for (int ii = 0; ii < 4; ++ii) a[ii] = load_a_rowmajor_f16(As[cur], TP, wm + 16 * ii, 0, lane);
#pragma unroll
        for (int j = 0; j < 2; ++j) b[j] = load_b_ncontig_k(Bs[cur], TP, wn + 16 * j, 0, lane);
#pragma unroll
        for (int ii = 0; ii < 4; ++ii)
#pragma unroll
            for (int j = 0; j < 2; ++j)
                acc[ii][j] = wmma16x16x32(a[ii], b[j], acc[ii][j]);
        __syncthreads();
    }

    int m0 = bm + wm, n0 = bn + wn;
#pragma unroll
    for (int ii = 0; ii < 4; ++ii)
#pragma unroll
        for (int j = 0; j < 2; ++j) {
            int col = n0 + 16 * j + (lane & 15);
            float bv = bias[col];
            int rb = m0 + 16 * ii + (lane >> 4) * 8;
#pragma unroll
            for (int r = 0; r < 8; ++r) {
                size_t idx = (size_t)(rb + r) * DIM + col;
                Y[idx] = acc[ii][j][r] + bv + resid[idx];
            }
        }
}

// One WG per (batch, 32-row q tile); loops over all 16 heads so the
// head-mean of probs is accumulated race-free by the owning WG.
// LDS: scores/probs tile 32 x 1032 f32 (padded stride -> no 64-bank conflicts).
#define SLDS 1032
__global__ __launch_bounds__(256)
void attention_kernel(const _Float16* __restrict__ Q, const _Float16* __restrict__ Km,
                      const _Float16* __restrict__ VT, _Float16* __restrict__ ctx,
                      float* __restrict__ attn_out) {
    extern __shared__ float s_sc[];              // 32 * 1032 floats = 132 KB
    int lane = threadIdx.x & 31;
    int wave = threadIdx.x >> 5;
    int b = blockIdx.x >> 4;
    int qbase = (blockIdx.x & 15) * 32;
    const _Float16* Qb = Q + (size_t)b * LQ * DIM;
    const _Float16* Kb = Km + (size_t)b * LK * DIM;
    const _Float16* VTb = VT + (size_t)b * DIM * LK;   // [d][kk]
    float* attn_b = attn_out + ((size_t)b * LQ + qbase) * LK;

    for (int h = 0; h < NH; ++h) {
        int hoff = h * HD;
        // ---- scores: S(32x1024) = Q_tile(32x64) @ K_h^T * (1/sqrt(HD))
        for (int t = wave; t < 32; t += 8) {
            int tm = t & 1, tg = t >> 1;
            v8f acc[4] = {};
#pragma unroll
            for (int k0 = 0; k0 < HD; k0 += 32) {
                v16h a = load_a_rowmajor_f16(Qb + hoff, DIM, qbase + tm * 16, k0, lane);
#pragma unroll
                for (int j = 0; j < 4; ++j) {
                    v16h bf = load_b_ncontig_k(Kb + hoff, DIM, tg * 64 + j * 16, k0, lane);
                    acc[j] = wmma16x16x32(a, bf, acc[j]);
                }
            }
            int rb = tm * 16 + (lane >> 4) * 8;
#pragma unroll
            for (int j = 0; j < 4; ++j) {
                int col = tg * 64 + j * 16 + (lane & 15);
#pragma unroll
                for (int r = 0; r < 8; ++r)
                    s_sc[(size_t)(rb + r) * SLDS + col] = acc[j][r] * 0.125f;
            }
        }
        __syncthreads();
        // ---- exact softmax, one wave per row
        for (int r = wave * 4; r < wave * 4 + 4; ++r) {
            float* row = s_sc + (size_t)r * SLDS;
            float mx = -3.0e38f;
            for (int k = lane; k < LK; k += 32) mx = fmaxf(mx, row[k]);
#pragma unroll
            for (int off = 16; off; off >>= 1) mx = fmaxf(mx, __shfl_xor(mx, off, 32));
            float sum = 0.f;
            for (int k = lane; k < LK; k += 32) {
                float e = __expf(row[k] - mx);
                row[k] = e;
                sum += e;
            }
#pragma unroll
            for (int off = 16; off; off >>= 1) sum += __shfl_xor(sum, off, 32);
            float inv = 1.f / sum;
            float* ar = attn_b + (size_t)r * LK;
            for (int k = lane; k < LK; k += 32) {
                float p = row[k] * inv;
                row[k] = p;
                float m = p * (1.f / (float)NH);
                if (h) m += ar[k];
                ar[k] = m;
            }
        }
        __syncthreads();
        // ---- ctx: C(32x64) = P(32x1024) @ V_h(1024x64); V^T -> contiguous B.
        {
            int tm = wave & 1, tn = wave >> 1;
            v8f acc = {};
            for (int k0 = 0; k0 < LK; k0 += 32) {
                v16h a  = load_a_lds_f32(s_sc, SLDS, tm * 16, k0, lane);
                v16h bf = load_b_ncontig_k(VTb + (size_t)hoff * LK, LK, tn * 16, k0, lane);
                acc = wmma16x16x32(a, bf, acc);
            }
            int col = tn * 16 + (lane & 15);
            int rb = qbase + tm * 16 + (lane >> 4) * 8;
#pragma unroll
            for (int r = 0; r < 8; ++r)
                ctx[((size_t)b * LQ + rb + r) * DIM + hoff + col] = (_Float16)acc[r];
        }
        __syncthreads();
    }
}

// Row-wise LayerNorm over DIM=1024, one block per row.
__global__ __launch_bounds__(256)
void layernorm_kernel(const float* __restrict__ X, const float* __restrict__ g,
                      const float* __restrict__ bt, float* __restrict__ Y) {
    __shared__ float red[256];
    size_t m = blockIdx.x;
    const float* x = X + m * DIM;
    float s = 0.f;
    for (int i = threadIdx.x; i < DIM; i += 256) s += x[i];
    red[threadIdx.x] = s;
    __syncthreads();
    for (int off = 128; off; off >>= 1) {
        if ((int)threadIdx.x < off) red[threadIdx.x] += red[threadIdx.x + off];
        __syncthreads();
    }
    float mu = red[0] * (1.f / DIM);
    __syncthreads();
    float v = 0.f;
    for (int i = threadIdx.x; i < DIM; i += 256) { float d = x[i] - mu; v += d * d; }
    red[threadIdx.x] = v;
    __syncthreads();
    for (int off = 128; off; off >>= 1) {
        if ((int)threadIdx.x < off) red[threadIdx.x] += red[threadIdx.x + off];
        __syncthreads();
    }
    float inv = rsqrtf(red[0] * (1.f / DIM) + EPS);
    for (int i = threadIdx.x; i < DIM; i += 256)
        Y[m * DIM + i] = (x[i] - mu) * inv * g[i] + bt[i];
}

// ---------------------------------------------------------------------------
extern "C" void kernel_launch(void* const* d_in, const int* in_sizes, int n_in,
                              void* d_out, int out_size, void* d_ws, size_t ws_size,
                              hipStream_t stream) {
    (void)in_sizes; (void)n_in; (void)out_size; (void)ws_size;
    const float* text   = (const float*)d_in[0];
    const float* vision = (const float*)d_in[1];
    const float* ipw    = (const float*)d_in[2];
    const float* ipb    = (const float*)d_in[3];
    const float* ow     = (const float*)d_in[4];
    const float* ob     = (const float*)d_in[5];
    const float* ln_g   = (const float*)d_in[6];
    const float* ln_b   = (const float*)d_in[7];

    float* out      = (float*)d_out;                       // (8,512,1024)
    float* attn_out = out + (size_t)NB * LQ * DIM;         // (8,512,1024)

    char* p = (char*)d_ws;
    _Float16* text16 = (_Float16*)p; p += (size_t)NB * LQ * DIM * 2;
    _Float16* vis16  = (_Float16*)p; p += (size_t)NB * LK * DIM * 2;
    _Float16* ipw16  = (_Float16*)p; p += (size_t)3 * DIM * DIM * 2;
    _Float16* ow16   = (_Float16*)p; p += (size_t)DIM * DIM * 2;
    _Float16* q16    = (_Float16*)p; p += (size_t)NB * LQ * DIM * 2;
    _Float16* k16    = (_Float16*)p; p += (size_t)NB * LK * DIM * 2;
    _Float16* v16    = (_Float16*)p; p += (size_t)NB * LK * DIM * 2;
    _Float16* vT16   = (_Float16*)p; p += (size_t)NB * DIM * LK * 2;
    _Float16* ctx16  = (_Float16*)p; p += (size_t)NB * LQ * DIM * 2;
    float*    att32  = (float*)p;

    const int MQ = NB * LQ;   // 4096
    const int MK = NB * LK;   // 8192

    // 1) f32 -> f16 conversions
    {
        int n;
        n = MQ * DIM;        cvt_f32_to_f16<<<(n + 255) / 256, 256, 0, stream>>>(text, text16, n);
        n = MK * DIM;        cvt_f32_to_f16<<<(n + 255) / 256, 256, 0, stream>>>(vision, vis16, n);
        n = 3 * DIM * DIM;   cvt_f32_to_f16<<<(n + 255) / 256, 256, 0, stream>>>(ipw, ipw16, n);
        n = DIM * DIM;       cvt_f32_to_f16<<<(n + 255) / 256, 256, 0, stream>>>(ow, ow16, n);
    }

    // 2) QKV projections via async-LDS double-buffered WMMA GEMM
    gemm_xwT_bias_f16<<<dim3(DIM / 128, MQ / 128), 256, 0, stream>>>(text16, ipw16, ipb, q16);
    gemm_xwT_bias_f16<<<dim3(DIM / 128, MK / 128), 256, 0, stream>>>(vis16, ipw16 + (size_t)DIM * DIM,
                                                                     ipb + DIM, k16);
    gemm_xwT_bias_f16<<<dim3(DIM / 128, MK / 128), 256, 0, stream>>>(vis16, ipw16 + (size_t)2 * DIM * DIM,
                                                                     ipb + 2 * DIM, v16);

    // 2b) transpose V so the P@V B-operand is K-contiguous
    transpose_v_kernel<<<dim3(DIM / 32, LK / 32, NB), 256, 0, stream>>>(v16, vT16);

    // 3) attention
    size_t attn_lds = (size_t)32 * SLDS * sizeof(float);   // 132 KB
    hipFuncSetAttribute((const void*)attention_kernel,
                        hipFuncAttributeMaxDynamicSharedMemorySize, (int)attn_lds);
    attention_kernel<<<NB * (LQ / 32), 256, attn_lds, stream>>>(q16, k16, vT16, ctx16, attn_out);

    // 4) output projection + bias + residual
    gemm_out_residual<<<dim3(DIM / 128, MQ / 128), 256, 0, stream>>>(ctx16, ow16, ob, text, att32);

    // 5) LayerNorm -> final output
    layernorm_kernel<<<MQ, 256, 0, stream>>>(att32, ln_g, ln_b, out);
}
